// EdgeConv_37881611551019
// MI455X (gfx1250) — compile-verified, matrix operand order checked by reading, not run on hardware
//
#include <hip/hip_runtime.h>
#include <hip/hip_bf16.h>

// ---------------------------------------------------------------------------
// EdgeConv for MI455X (gfx1250, wave32, WMMA)
// ---------------------------------------------------------------------------

#define BB   8
#define NN   8192
#define CC   64
#define PP   2048
#define KK   16
#define CIN  134          // 2*C + 6
#define H1C  64
#define H2C  128
#define FXS  68           // padded feat_x row stride (67 -> 68 floats)
#define EPSV 1e-5f

typedef __bf16 bf16;
typedef bf16  v16bf __attribute__((ext_vector_type(16)));
typedef bf16  v8bf  __attribute__((ext_vector_type(8)));
typedef float v8f   __attribute__((ext_vector_type(8)));
typedef unsigned int v4u __attribute__((ext_vector_type(4)));
typedef int   v8i   __attribute__((ext_vector_type(8)));
typedef int   v4i   __attribute__((ext_vector_type(4)));

#if defined(__has_builtin)
#  if __has_builtin(__builtin_amdgcn_tensor_load_to_lds) && __has_builtin(__builtin_amdgcn_s_wait_tensorcnt)
#    define HAVE_TDM 1
#  endif
#endif
#ifndef HAVE_TDM
#  define HAVE_TDM 0
#endif

// ---------------------------------------------------------------------------
// Kernel 1: build feat_x rows (B,N,68) = [feat^T | xyz | pad], and emit
// new_xyz / sample_idx outputs.
// ---------------------------------------------------------------------------
__global__ __launch_bounds__(256) void featx_build(
    const float* __restrict__ xyz, const float* __restrict__ feat,
    float* __restrict__ featx, float* __restrict__ new_xyz,
    int* __restrict__ sample_idx)
{
  const int b = blockIdx.y;
  const int n = blockIdx.x * 256 + threadIdx.x;
  const float* fb = feat + (size_t)b * CC * NN;
  float* row = featx + ((size_t)b * NN + n) * FXS;
#pragma unroll 8
  for (int c = 0; c < CC; ++c) row[c] = fb[(size_t)c * NN + n];   // coalesced reads
  const float* x3 = xyz + ((size_t)b * NN + n) * 3;
  const float px = x3[0], py = x3[1], pz = x3[2];
  row[64] = px; row[65] = py; row[66] = pz; row[67] = 0.f;
  if (n < PP) {
    float* nx = new_xyz + ((size_t)b * PP + n) * 3;
    nx[0] = px; nx[1] = py; nx[2] = pz;
    sample_idx[b * PP + n] = n;
  }
}

// ---------------------------------------------------------------------------
// Kernel 2: kNN (K=16 smallest d2). One wave per query point.
// xyz for the batch (96 KB) staged into LDS with the Tensor Data Mover.
// Dynamic LDS: [0,96K) xyz ; [96K,112K) cand d2 ; [112K,128K) cand idx
// ---------------------------------------------------------------------------
__global__ __launch_bounds__(256) void knn_topk(
    const float* __restrict__ xyz, int* __restrict__ knn)
{
  extern __shared__ char smem[];
  float* xs = (float*)smem;                       // 24576 floats
  float* cd = (float*)(smem + 98304);             // 8 waves * 512
  int*   ci = (int*)(smem + 98304 + 16384);

  const int b = blockIdx.y;
  const float* xb = xyz + (size_t)b * NN * 3;

#if HAVE_TDM
  if (threadIdx.x < 32) {
    // Build Tensor DMA descriptor: 1-D tile of 24576 f32 elements -> LDS.
    unsigned lds_off = (unsigned)(unsigned long long)xs;      // low 32 bits = LDS offset
    unsigned long long ga = (unsigned long long)xb;
    v4u g0;
    g0[0] = 1u;                                   // count=1, is_restore=0, gather=0
    g0[1] = lds_off;                              // lds_addr (bytes)
    g0[2] = (unsigned)(ga & 0xFFFFFFFFull);       // global_addr[31:0]
    g0[3] = (unsigned)((ga >> 32) & 0x01FFFFFFull) | (2u << 30);  // addr[56:32] | type=2
    v8i g1;
    g1[0] = (int)(2u << 16);                      // wg_mask=0, data_size=2 (4B)
    g1[1] = (int)(24576u << 16);                  // tensor_dim0[15:0] in bits[63:48]
    g1[2] = (int)(1u << 16);                      // tensor_dim0 hi=0 ; tensor_dim1=1 (lo16)
    g1[3] = (int)(24576u << 16);                  // tensor_dim1 hi=0 ; tile_dim0=24576
    g1[4] = 1;                                    // tile_dim1=1, tile_dim2=0
    g1[5] = 24576;                                // tensor_dim0_stride (lo32)
    g1[6] = 0;
    g1[7] = 0;
    v4i z4 = {};
#if __clang_major__ >= 23
    v8i z8 = {};
    __builtin_amdgcn_tensor_load_to_lds(g0, g1, z4, z4, z8, 0);
#else
    __builtin_amdgcn_tensor_load_to_lds(g0, g1, z4, z4, 0);
#endif
    __builtin_amdgcn_s_wait_tensorcnt(0);         // s_wait_tensorcnt 0
  }
#else
  for (int i = threadIdx.x; i < NN * 3; i += 256) xs[i] = xb[i];
#endif
  __syncthreads();

  const int wave = threadIdx.x >> 5;
  const int lane = threadIdx.x & 31;
  const int p = blockIdx.x * 8 + wave;

  const float qx = xs[p * 3 + 0], qy = xs[p * 3 + 1], qz = xs[p * 3 + 2];

  // Per-lane sorted (ascending) top-16 in registers; fully unrolled insert.
  float bd[16]; int bi[16];
#pragma unroll
  for (int j = 0; j < 16; ++j) { bd[j] = 3.4e38f; bi[j] = 0; }

  for (int n0 = lane; n0 < NN; n0 += 32) {
    const float dx = xs[n0 * 3 + 0] - qx;
    const float dy = xs[n0 * 3 + 1] - qy;
    const float dz = xs[n0 * 3 + 2] - qz;
    const float d = dx * dx + dy * dy + dz * dz;
    if (d < bd[15]) {
      float cdv = d; int civ = n0;
#pragma unroll
      for (int j = 0; j < 16; ++j) {
        if (bd[j] > cdv) {
          float td = bd[j]; bd[j] = cdv; cdv = td;
          int   ti = bi[j]; bi[j] = civ; civ = ti;
        }
      }
    }
  }

  // Dump sorted lists, then 32-way merge (16 rounds of wave argmin).
  float* wd = cd + wave * 512;
  int*   wi = ci + wave * 512;
#pragma unroll
  for (int j = 0; j < 16; ++j) { wd[lane * 16 + j] = bd[j]; wi[lane * 16 + j] = bi[j]; }
  __syncthreads();

  int ptr = 0;
  int* outp = knn + ((size_t)b * PP + p) * KK;
  for (int r = 0; r < KK; ++r) {
    float hv = (ptr < 16) ? wd[lane * 16 + ptr] : 3.4e38f;
    int   hi = (ptr < 16) ? wi[lane * 16 + ptr] : 0;
    int   hl = lane;
#pragma unroll
    for (int off = 16; off > 0; off >>= 1) {
      const float ov = __shfl_xor(hv, off, 32);
      const int   oi = __shfl_xor(hi, off, 32);
      const int   ol = __shfl_xor(hl, off, 32);
      if (ov < hv || (ov == hv && ol < hl)) { hv = ov; hi = oi; hl = ol; }
    }
    if (lane == 0) outp[r] = hi;
    if (lane == hl) ++ptr;
  }
}

// ---------------------------------------------------------------------------
// Kernel 3: gather + edge features + 2-layer MLP (WMMA bf16) + max over K.
// One wave per query point (16 edge rows == one 16-row M-tile).
// W tiles pre-swizzled into WMMA-B lane layout in LDS.
// ---------------------------------------------------------------------------
__global__ __launch_bounds__(256) void edge_mlp(
    const float* __restrict__ featx, const int* __restrict__ knn,
    const float* __restrict__ W1, const float* __restrict__ b1,
    const float* __restrict__ g1, const float* __restrict__ be1,
    const float* __restrict__ m1, const float* __restrict__ v1,
    const float* __restrict__ W2, const float* __restrict__ b2,
    const float* __restrict__ g2, const float* __restrict__ be2,
    const float* __restrict__ m2, const float* __restrict__ v2,
    float* __restrict__ out_feat)
{
  __shared__ bf16 w1s[20 * 512];      // 5 K-tiles x 4 N-tiles, B-layout
  __shared__ bf16 w2s[16 * 512];      // 2 K-tiles x 8 N-tiles, B-layout
  __shared__ float s1[H1C], t1[H1C], s2[H2C], t2[H2C];
  __shared__ bf16 hls[8 * 16 * 64];   // per-wave 16x64 bf16 h1 scratch

  const int tid = threadIdx.x;

  // Stage W1 in WMMA B layout: lane l, element e of a 32x16 tile holds
  // K = kt*32 + e + (l>>4)*8 + (e>=8 ? 8 : 0), N = nt*16 + (l&15).
  for (int i = tid; i < 20 * 512; i += 256) {
    const int t = i >> 9, r = i & 511, l = r >> 4, e = r & 15;
    const int kt = t >> 2, nt = t & 3;
    const int ke = e + (l >> 4) * 8 + (e >= 8 ? 8 : 0);
    const int k = kt * 32 + ke;
    const int n = nt * 16 + (l & 15);
    w1s[i] = (bf16)((k < CIN) ? W1[n * CIN + k] : 0.f);
  }
  for (int i = tid; i < 16 * 512; i += 256) {
    const int t = i >> 9, r = i & 511, l = r >> 4, e = r & 15;
    const int kt = t >> 3, nt = t & 7;
    const int ke = e + (l >> 4) * 8 + (e >= 8 ? 8 : 0);
    const int k = kt * 32 + ke;
    const int n = nt * 16 + (l & 15);
    w2s[i] = (bf16)W2[n * H1C + k];
  }
  if (tid < H1C) {
    const float s = g1[tid] * rsqrtf(v1[tid] + EPSV);
    s1[tid] = s; t1[tid] = s * (b1[tid] - m1[tid]) + be1[tid];
  }
  if (tid < H2C) {
    const float s = g2[tid] * rsqrtf(v2[tid] + EPSV);
    s2[tid] = s; t2[tid] = s * (b2[tid] - m2[tid]) + be2[tid];
  }
  __syncthreads();

  const int wave = tid >> 5, lane = tid & 31;
  const int half = lane >> 4, col = lane & 15;
  const int b = blockIdx.y;
  const int p = blockIdx.x * 8 + wave;
  const int m = col;                              // A-matrix row owned by this lane

  const int nidx = knn[((size_t)b * PP + p) * KK + m];
  const float* nf = featx + ((size_t)b * NN + nidx) * FXS;
  const float* cf = featx + ((size_t)b * NN + p) * FXS;

  // ----- Layer 1: (16 x 160pad) * (160pad x 64) -----
  v8f acc1[4] = {};
#pragma unroll
  for (int kt = 0; kt < 5; ++kt) {
    v16bf a;
#pragma unroll
    for (int e = 0; e < 16; ++e) {
      const int ke = e + half * 8 + (e >= 8 ? 8 : 0);
      const int c = kt * 32 + ke;
      // Branch-free gather: unconditional clamped loads + v_cndmask selects.
      const int cn  = (c < 66) ? c : 66;                  // clamp for nf
      const int ccr = (c < 67) ? c : (c - 67);            // center index
      const int cc  = (ccr < 66) ? ccr : 66;              // clamp for cf
      const float vn = nf[cn];
      const float vc = cf[cc];
      const float x = (c < 67) ? (vn - vc) : ((c < CIN) ? vc : 0.f);
      a[e] = (bf16)x;
    }
#pragma unroll
    for (int nt = 0; nt < 4; ++nt) {
      const v16bf bv = *(const v16bf*)&w1s[(kt * 4 + nt) * 512 + lane * 16];
      acc1[nt] = __builtin_amdgcn_wmma_f32_16x16x32_bf16(
          false, a, false, bv, (short)0, acc1[nt], false, false);
    }
  }

  // BN + ReLU, store h1 (C/D layout -> row-major bf16 scratch)
  bf16* hw = hls + wave * 1024;
#pragma unroll
  for (int nt = 0; nt < 4; ++nt) {
    const int o = nt * 16 + col;
    const float sc = s1[o], tc = t1[o];
#pragma unroll
    for (int v = 0; v < 8; ++v) {
      const float h = fmaxf(0.f, sc * acc1[nt][v] + tc);
      hw[(half * 8 + v) * 64 + o] = (bf16)h;      // row M = half*8 + v
    }
  }
  __syncthreads();

  // ----- Layer 2: (16 x 64) * (64 x 128) -----
  v8f acc2[8] = {};
#pragma unroll
  for (int kt = 0; kt < 2; ++kt) {
    const bf16* base = hw + m * 64 + kt * 32 + half * 8;
    const v8bf lo = *(const v8bf*)base;           // K = kt*32 + half*8 + [0..7]
    const v8bf hi = *(const v8bf*)(base + 16);    // K = kt*32 + half*8 + 16 + [0..7]
    const v16bf a2 = __builtin_shufflevector(lo, hi,
        0, 1, 2, 3, 4, 5, 6, 7, 8, 9, 10, 11, 12, 13, 14, 15);
#pragma unroll
    for (int nt = 0; nt < 8; ++nt) {
      const v16bf bv = *(const v16bf*)&w2s[(kt * 8 + nt) * 512 + lane * 16];
      acc2[nt] = __builtin_amdgcn_wmma_f32_16x16x32_bf16(
          false, a2, false, bv, (short)0, acc2[nt], false, false);
    }
  }

  // BN + ReLU + max over K (16 rows): 8 rows per lane, then shfl_xor(16).
#pragma unroll
  for (int nt = 0; nt < 8; ++nt) {
    const int o = nt * 16 + col;
    const float sc = s2[o], tc = t2[o];
    float pm = 0.f;                               // ReLU outputs are >= 0
#pragma unroll
    for (int v = 0; v < 8; ++v)
      pm = fmaxf(pm, fmaxf(0.f, sc * acc2[nt][v] + tc));
    const float om = __shfl_xor(pm, 16, 32);
    pm = fmaxf(pm, om);
    if (half == 0)
      out_feat[(size_t)b * H2C * PP + (size_t)o * PP + p] = pm;
  }
}

// ---------------------------------------------------------------------------
// Launch
// ---------------------------------------------------------------------------
extern "C" void kernel_launch(void* const* d_in, const int* in_sizes, int n_in,
                              void* d_out, int out_size, void* d_ws, size_t ws_size,
                              hipStream_t stream) {
  (void)in_sizes; (void)n_in; (void)out_size; (void)ws_size;
  const float* xyz  = (const float*)d_in[0];
  const float* feat = (const float*)d_in[1];
  const float* W1   = (const float*)d_in[2];
  const float* b1   = (const float*)d_in[3];
  const float* g1   = (const float*)d_in[4];
  const float* be1  = (const float*)d_in[5];
  const float* m1   = (const float*)d_in[6];
  const float* v1   = (const float*)d_in[7];
  const float* W2   = (const float*)d_in[8];
  const float* b2   = (const float*)d_in[9];
  const float* g2   = (const float*)d_in[10];
  const float* be2  = (const float*)d_in[11];
  const float* m2   = (const float*)d_in[12];
  const float* v2   = (const float*)d_in[13];

  float* out      = (float*)d_out;
  float* new_xyz  = out;                                    // B*P*3
  float* new_feat = out + (size_t)BB * PP * 3;              // B*128*P
  int*   samp     = (int*)(out + (size_t)BB * PP * 3 + (size_t)BB * H2C * PP);

  float* featx = (float*)d_ws;                              // B*N*68 f32
  int*   knn   = (int*)((char*)d_ws + (size_t)BB * NN * FXS * sizeof(float));

  featx_build<<<dim3(NN / 256, BB), 256, 0, stream>>>(xyz, feat, featx, new_xyz, samp);
  knn_topk  <<<dim3(PP / 8, BB), 256, 131072, stream>>>(xyz, knn);
  edge_mlp  <<<dim3(PP / 8, BB), 256, 0, stream>>>(featx, knn,
      W1, b1, g1, be1, m1, v1, W2, b2, g2, be2, m2, v2, new_feat);
}
